// StyleBlock2_6107443495260
// MI455X (gfx1250) — compile-verified, matrix operand order checked by reading, not run on hardware
//
#include <hip/hip_runtime.h>

typedef __attribute__((ext_vector_type(16))) __bf16       v16bf;
typedef __attribute__((ext_vector_type(4)))  __bf16       v4bf;
typedef __attribute__((ext_vector_type(8)))  float        v8f;
typedef __attribute__((ext_vector_type(4)))  unsigned int v4u;

#define NB    8
#define HH    64
#define WW    64
#define CINC  512
#define COUTC 512
#define ZD    512
#define KTOT  4608            // 9*512
#define NCHUNK 144            // KTOT/32
#define DCOEF 0.0625f         // sqrt(2/512) exactly
#define CCOEF (1.0f/48.0f)    // sqrt(2/4608) exactly

// workspace layout (bytes, 256-aligned)
#define OFF_S    0u            // s      : 8*512 f32           (16 KB)
#define OFF_D    16384u        // d      : 8*512 f32           (16 KB)
#define OFF_W2   32768u        // w2     : 512*512 f32         (1 MB)
#define OFF_WT   1081344u      // B swizzled bf16              (4.5 MB)
#define OFF_XPAD 5799936u      // padded x bf16 8*66*66*512    (34 MB)

union Frag16 { v16bf v; v4u q[2]; };

// ---- s[n][c] = latent[n,:] . (dense_w[:,c]*DCOEF) + dense_b[c] -------------
__global__ __launch_bounds__(256) void k_style(const float* __restrict__ latent,
                                               const float* __restrict__ dw,
                                               const float* __restrict__ db,
                                               float* __restrict__ s) {
  int idx = blockIdx.x * 256 + threadIdx.x;     // 4096
  int c = idx & 511, n = idx >> 9;
  float acc = 0.f;
  for (int z = 0; z < ZD; ++z)
    acc += latent[(n << 9) + z] * dw[(z << 9) + c];
  s[idx] = acc * DCOEF + db[c];
}

// ---- w2[cin][cout] = sum_k (w*CCOEF)^2 -------------------------------------
__global__ __launch_bounds__(256) void k_w2(const float* __restrict__ cw,
                                            float* __restrict__ w2) {
  int idx = blockIdx.x * 256 + threadIdx.x;     // 262144
  int cout = idx & 511, cin = idx >> 9;
  float acc = 0.f;
  for (int kk = 0; kk < 9; ++kk) {
    float v = cw[(size_t)((kk << 9) + cin) * COUTC + cout] * CCOEF;
    acc += v * v;
  }
  w2[idx] = acc;                                 // [cin][cout]
}

// ---- d[n][cout] = rsqrt( (s^2) . w2 + 1e-8 ) -------------------------------
__global__ __launch_bounds__(256) void k_demod(const float* __restrict__ s,
                                               const float* __restrict__ w2,
                                               float* __restrict__ dv) {
  int idx = blockIdx.x * 256 + threadIdx.x;     // 4096
  int cout = idx & 511, n = idx >> 9;
  float acc = 0.f;
  for (int ci = 0; ci < CINC; ++ci) {
    float sv = s[(n << 9) + ci];
    acc += sv * sv * w2[(ci << 9) + cout];
  }
  dv[idx] = rsqrtf(acc + 1e-8f);
}

// ---- swizzle B into per-lane WMMA fragment layout --------------------------
// element at ((ck*32 + nT)*32 + lane)*16 + f  holds  w[kh][kw][cin][col]*CCOEF
__global__ __launch_bounds__(256) void k_prepB(const float* __restrict__ cw,
                                               __bf16* __restrict__ wsw) {
  int idx = blockIdx.x * 256 + threadIdx.x;     // 2359296
  int f  = idx & 15;
  int t  = (idx >> 4) & 31;
  int nT = (idx >> 9) & 31;
  int ck = idx >> 14;                           // 0..143
  int base = (t >> 4) << 3;                     // 0 or 8
  int koff = (f < 8) ? (base + f) : (base + 16 + (f - 8));
  int k    = ck * 32 + koff;
  int kk   = k >> 9;
  int cin  = k & 511;
  int col  = (nT << 4) + (t & 15);
  float v = cw[(size_t)((kk << 9) + cin) * COUTC + col] * CCOEF;
  wsw[idx] = (__bf16)v;
}

// ---- zero padded buffer (halo must be 0; ws is poisoned 0xAA) --------------
__global__ __launch_bounds__(256) void k_zero(__bf16* __restrict__ xpad) {
  size_t i = ((size_t)blockIdx.x * 256 + threadIdx.x) * 4;   // 17842176 total
  *(unsigned long long*)(xpad + i) = 0ull;
}

// ---- xpad[n][h+1][w+1][c] = bf16(data * s), 4 channels per thread ----------
__global__ __launch_bounds__(256) void k_mod(const float* __restrict__ data,
                                             const float* __restrict__ s,
                                             __bf16* __restrict__ xpad) {
  size_t base = ((size_t)blockIdx.x * 256 + threadIdx.x) * 4;  // < 16777216
  int cin = (int)(base & 511);
  int w   = (int)((base >> 9) & 63);
  int h   = (int)((base >> 15) & 63);
  int n   = (int)(base >> 21);
  float4 v  = *(const float4*)(data + base);
  float4 sv = *(const float4*)(s + (n << 9) + cin);
  v4bf o;
  o[0] = (__bf16)(v.x * sv.x);
  o[1] = (__bf16)(v.y * sv.y);
  o[2] = (__bf16)(v.z * sv.z);
  o[3] = (__bf16)(v.w * sv.w);
  size_t dst = ((size_t)((n * 66 + h + 1) * 66) + (w + 1)) * 512 + cin;
  *(v4bf*)(xpad + dst) = o;
}

// ---- implicit-GEMM conv: LDS double-buffered bf16 WMMA + fused epilogue ----
// grid (256, 4), block 256 = 8 waves arranged 4(M) x 2(N).
// Block tile 128M x 128N; wave: 2 M-tiles x 4 N-tiles (8 WMMA / chunk).
__global__ __launch_bounds__(256) void k_conv(const __bf16* __restrict__ xpad,
                                              const __bf16* __restrict__ wsw,
                                              const float* __restrict__ dv,
                                              const float* __restrict__ bias,
                                              const float* __restrict__ ncoef,
                                              const float* __restrict__ noise,
                                              float* __restrict__ out) {
  // A: [kseg 0..3][row 0..127] 16B units ; B: [tile 0..7][half 0..1][lane] 16B
  __shared__ v4u ldsA[2][512];
  __shared__ v4u ldsB[2][512];

  const int tid  = threadIdx.x;
  const int lane = tid & 31;
  const int wave = tid >> 5;
  const int mSub = wave >> 1;                   // 0..3
  const int nSub = wave & 1;                    // 0..1

  // ---- fill-role addressing (all 256 threads) ----
  const int fR    = tid >> 1;                   // A row in block, 0..127
  const int fHalf = tid & 1;                    // which 32B half of the row
  const int fM = blockIdx.x * 128 + fR;
  const int fN = fM >> 12, fH = (fM >> 6) & 63, fW = fM & 63;
  const __bf16* aFillBase =
      xpad + ((size_t)((fN * 66 + fH + 1) * 66) + (fW + 1)) * 512 + fHalf * 16;
  const __bf16* bFillBase =
      wsw + ((size_t)((blockIdx.y * 8 + wave) * 32 + lane)) * 16;
  const int aDst0 = (fHalf * 2) * 128 + fR;     // v4u index (seg, seg+1)
  const int bDst0 = wave * 64 + lane;           // v4u index (half0, half0+32)

  // ---- compute-role addressing ----
  const int rowL = mSub * 32 + (lane & 15);     // + i*16 per M-tile
  const int aSeg = lane >> 4;                   // 0/1
  v8f acc[2][4] = {};

  // ---- prologue: fill buffer 0 with chunk 0 (kk=0 -> tap (-1,-1), c0=0) ----
  {
    const __bf16* ap = aFillBase + (-67) * 512;
    const v4u fa0 = *(const v4u*)ap;
    const v4u fa1 = *(const v4u*)(ap + 8);
    const v4u fb0 = *(const v4u*)bFillBase;
    const v4u fb1 = *(const v4u*)(bFillBase + 8);
    ldsA[0][aDst0]       = fa0;
    ldsA[0][aDst0 + 128] = fa1;
    ldsB[0][bDst0]       = fb0;
    ldsB[0][bDst0 + 32]  = fb1;
  }
  __syncthreads();

  for (int ck = 0; ck < NCHUNK; ++ck) {
    const int cur = ck & 1;
    const bool more = (ck + 1) < NCHUNK;
    v4u fa0, fa1, fb0, fb1;
    if (more) {                                  // prefetch chunk ck+1
      const int ckn = ck + 1;
      const int kk = ckn >> 4;
      const int c0 = (ckn & 15) << 5;
      const int tap = ((kk / 3 - 1) * 66 + (kk % 3 - 1)) * 512;
      const __bf16* ap = aFillBase + tap + c0;
      fa0 = *(const v4u*)ap;
      fa1 = *(const v4u*)(ap + 8);
      const __bf16* bp = bFillBase + (size_t)ckn * 16384;
      fb0 = *(const v4u*)bp;
      fb1 = *(const v4u*)(bp + 8);
    }

    // ---- compute chunk ck from LDS: load ALL fragments into dedicated regs,
    //      then run the 8 WMMAs back-to-back (no mid-stream dscnt==0 stalls).
    Frag16 a0, a1, b[4];
    a0.q[0] = ldsA[cur][aSeg * 128 + rowL];
    a0.q[1] = ldsA[cur][(aSeg + 2) * 128 + rowL];
    a1.q[0] = ldsA[cur][aSeg * 128 + rowL + 16];
    a1.q[1] = ldsA[cur][(aSeg + 2) * 128 + rowL + 16];
#pragma unroll
    for (int j = 0; j < 4; ++j) {
      const int t = nSub * 4 + j;
      b[j].q[0] = ldsB[cur][t * 64 + lane];
      b[j].q[1] = ldsB[cur][t * 64 + 32 + lane];
    }
#pragma unroll
    for (int j = 0; j < 4; ++j) {
      acc[0][j] = __builtin_amdgcn_wmma_f32_16x16x32_bf16(
          false, a0.v, false, b[j].v, (short)0, acc[0][j], false, false);
      acc[1][j] = __builtin_amdgcn_wmma_f32_16x16x32_bf16(
          false, a1.v, false, b[j].v, (short)0, acc[1][j], false, false);
    }

    if (more) {                                  // commit prefetch to LDS
      const int nxt = cur ^ 1;
      ldsA[nxt][aDst0]       = fa0;
      ldsA[nxt][aDst0 + 128] = fa1;
      ldsB[nxt][bDst0]       = fb0;
      ldsB[nxt][bDst0 + 32]  = fb1;
    }
    __syncthreads();
  }

  // ---- epilogue: y = conv*d + bias + noise*ncoef ; leaky relu 0.2 ----------
  const int colL = lane & 15;
  const int rowHalf = (lane >> 4) << 3;
#pragma unroll
  for (int i = 0; i < 2; ++i) {
    const int rowBase = blockIdx.x * 128 + mSub * 32 + i * 16 + rowHalf;
    const int nimg = rowBase >> 12;
    const float* dn = dv + (nimg << 9);
#pragma unroll
    for (int r = 0; r < 8; ++r) {
      const size_t ob = (size_t)(rowBase + r) * COUTC;
#pragma unroll
      for (int j = 0; j < 4; ++j) {
        const int col = blockIdx.y * 128 + nSub * 64 + j * 16 + colL;
        float y = acc[i][j][r] * dn[col] + bias[col] + noise[ob + col] * ncoef[col];
        out[ob + col] = y >= 0.f ? y : 0.2f * y;
      }
    }
  }
}

extern "C" void kernel_launch(void* const* d_in, const int* in_sizes, int n_in,
                              void* d_out, int out_size, void* d_ws, size_t ws_size,
                              hipStream_t stream) {
  const float* data    = (const float*)d_in[0];
  const float* latent  = (const float*)d_in[1];
  const float* dense_w = (const float*)d_in[2];
  const float* dense_b = (const float*)d_in[3];
  const float* conv_w  = (const float*)d_in[4];
  const float* bias    = (const float*)d_in[5];
  const float* ncoef   = (const float*)d_in[6];
  const float* noise   = (const float*)d_in[7];
  float* out = (float*)d_out;

  char* ws = (char*)d_ws;
  float*  s    = (float*)(ws + OFF_S);
  float*  dv   = (float*)(ws + OFF_D);
  float*  w2   = (float*)(ws + OFF_W2);
  __bf16* wsw  = (__bf16*)(ws + OFF_WT);
  __bf16* xpad = (__bf16*)(ws + OFF_XPAD);

  k_style<<<16, 256, 0, stream>>>(latent, dense_w, dense_b, s);
  k_w2<<<1024, 256, 0, stream>>>(conv_w, w2);
  k_prepB<<<9216, 256, 0, stream>>>(conv_w, wsw);
  k_demod<<<16, 256, 0, stream>>>(s, w2, dv);
  k_zero<<<17424, 256, 0, stream>>>(xpad);      // 8*66*66*512 elements
  k_mod<<<16384, 256, 0, stream>>>(data, s, xpad);

  dim3 grid(256, 4);
  k_conv<<<grid, 256, 0, stream>>>(xpad, wsw, dv, bias, ncoef, noise, out);
}
